// ChannelSA_19035295056394
// MI455X (gfx1250) — matrix operand order repaired and checked by reading, not compiled, not run on hardware
//
#include <hip/hip_runtime.h>
#include <hip/hip_bf16.h>

// ---------------------------------------------------------------------------
// ChannelSA for MI455X (gfx1250, wave32, WMMA).
//   K1: qkv_pre = W1 @ x           (f16 WMMA GEMM, fp32 accum)
//   K2: qkv = dwconv3x3(qkv_pre)   (float4 VALU, bandwidth bound)
//   K3: inv L2 norms of q,k rows over L=16384
//   K4: attn_raw += qhat @ khat^T  (f16 WMMA, K-split + ds_add + global atomics)
//   K5: softmax(attn/sqrt(48)+pos); M[b] = W_out @ blockdiag(attn[b])
//   K6: out = M[b] @ v[b]          (same WMMA GEMM kernel)
// ---------------------------------------------------------------------------

typedef __attribute__((ext_vector_type(16))) _Float16 v16h;
typedef __attribute__((ext_vector_type(2)))  __fp16   v2fp16;
typedef __attribute__((ext_vector_type(8)))  float    v8f;

union AFrag { v16h h; unsigned int u[8]; };

// packed f32x2 -> f16x2 (v_cvt_pk_rtz_f16_f32), with scale
static __device__ __forceinline__ unsigned int pk2(float x, float y, float s) {
    union { v2fp16 h; unsigned int u; } t;
    t.h = __builtin_amdgcn_cvt_pkrtz(x * s, y * s);
    return t.u;
}

// ---- A fragment (16xK=32 f16) from LDS tile stored [m][k] (K-contiguous) ----
// lanes 0-15 M=lane, VGPR j holds K = 2j (+1);
//             lanes 16-31 same with K base +16.
static __device__ __forceinline__ void load_a_frag_lds(const _Float16* As, int mbase,
                                                       int lane, AFrag& f) {
    int m  = mbase + (lane & 15);
    int kb = (lane < 16) ? 0 : 16;
    const unsigned int* row = (const unsigned int*)(As + m * 32);
#pragma unroll
    for (int j = 0; j < 8; ++j) {
        f.u[j] = row[(kb >> 1) + j];
    }
}

// ---- B fragment (K=32 x 16 f16) from LDS tile stored transposed [n][k] ----
// lanes 0-15: N=lane, K=0..15 (VGPR j -> K=2j,2j+1); lanes 16-31: K=16..31.
static __device__ __forceinline__ void load_b_frag_lds(const _Float16* Bs, int nbase,
                                                       int lane, AFrag& f) {
    int n  = nbase + (lane & 15);
    int kb = (lane < 16) ? 0 : 16;
    const unsigned int* row = (const unsigned int*)(Bs + n * 32);
#pragma unroll
    for (int j = 0; j < 8; ++j) f.u[j] = row[(kb >> 1) + j];
}

// ===========================================================================
// K1/K6: fp32-in/fp32-out GEMM via f16 WMMA.  C[M,N] = A[M,K] x B[K,N].
// BM=64, BN=128, BK=32; 256 threads = 8 waves (wm 0..1, wn 0..3),
// each wave computes a 32x32 tile -> 4 WMMAs per k-step.
// Stager: issue all 6 b128 global loads first (MLP), then packed-convert;
// B transpose stages k-row pairs so every LDS store is a packed b32.
// ===========================================================================
__global__ __launch_bounds__(256)
void gemm_f16wmma(const float* __restrict__ A, long aStride,
                  const float* __restrict__ B, long bStride,
                  float* __restrict__ C, long cStride,
                  int M, int N, int K) {
    constexpr int BM = 64, BN = 128, BK = 32;
    __shared__ __align__(16) _Float16 As[BM * BK];   // [m][k]
    __shared__ __align__(16) _Float16 Bs[BN * BK];   // [n][k] (transposed)

    const int tid  = threadIdx.x;
    const int lane = tid & 31;
    const int wave = tid >> 5;
    const int wm = wave >> 2, wn = wave & 3;
    const int m0 = blockIdx.y * BM;
    const int n0 = blockIdx.x * BN;
    const long bz = blockIdx.z;

    const float* Ab = A + bz * aStride;
    const float* Bb = B + bz * bStride;
    float*       Cb = C + bz * cStride;

    // per-thread stager coordinates (constant across k-steps)
    const int mA0 = tid >> 3,           kA0 = (tid & 7) << 2;          // A quad 0
    const int mA1 = (tid + 256) >> 3,   kA1 = kA0;                     // A quad 1
    const int kB0 = (tid >> 5) << 1,    nB0 = (tid & 31) << 2;         // B row-pair 0
    const int kB1 = ((tid + 256) >> 5) << 1, nB1 = nB0;                // B row-pair 1

    v8f acc[2][2] = {{{}, {}}, {{}, {}}};

    for (int k0 = 0; k0 < K; k0 += BK) {
        // ---- issue all global loads up front (6x b128 in flight) ----
        float4 a0 = *(const float4*)(Ab + (long)(m0 + mA0) * K + (k0 + kA0));
        float4 a1 = *(const float4*)(Ab + (long)(m0 + mA1) * K + (k0 + kA1));
        float4 b00 = *(const float4*)(Bb + (long)(k0 + kB0)     * N + (n0 + nB0));
        float4 b01 = *(const float4*)(Bb + (long)(k0 + kB0 + 1) * N + (n0 + nB0));
        float4 b10 = *(const float4*)(Bb + (long)(k0 + kB1)     * N + (n0 + nB1));
        float4 b11 = *(const float4*)(Bb + (long)(k0 + kB1 + 1) * N + (n0 + nB1));
        if (k0 + BK < K) {  // global_prefetch of next tiles
            __builtin_prefetch(Bb + (long)(k0 + BK + (tid >> 3)) * N + n0 + ((tid & 7) << 4), 0, 1);
            __builtin_prefetch(Ab + (long)(m0 + (tid >> 2)) * K + k0 + BK, 0, 1);
        }

        // ---- convert + store to LDS ----
        *(uint2*)(As + mA0 * BK + kA0) = uint2{pk2(a0.x, a0.y, 1.f), pk2(a0.z, a0.w, 1.f)};
        *(uint2*)(As + mA1 * BK + kA1) = uint2{pk2(a1.x, a1.y, 1.f), pk2(a1.z, a1.w, 1.f)};
        // B: pack (k, k+1) pairs -> one b32 per column, layout [n][k]
        *(unsigned int*)(Bs + (nB0 + 0) * BK + kB0) = pk2(b00.x, b01.x, 1.f);
        *(unsigned int*)(Bs + (nB0 + 1) * BK + kB0) = pk2(b00.y, b01.y, 1.f);
        *(unsigned int*)(Bs + (nB0 + 2) * BK + kB0) = pk2(b00.z, b01.z, 1.f);
        *(unsigned int*)(Bs + (nB0 + 3) * BK + kB0) = pk2(b00.w, b01.w, 1.f);
        *(unsigned int*)(Bs + (nB1 + 0) * BK + kB1) = pk2(b10.x, b11.x, 1.f);
        *(unsigned int*)(Bs + (nB1 + 1) * BK + kB1) = pk2(b10.y, b11.y, 1.f);
        *(unsigned int*)(Bs + (nB1 + 2) * BK + kB1) = pk2(b10.z, b11.z, 1.f);
        *(unsigned int*)(Bs + (nB1 + 3) * BK + kB1) = pk2(b10.w, b11.w, 1.f);
        __syncthreads();

        AFrag fa0, fa1, fb0, fb1;
        load_a_frag_lds(As, wm * 32,      lane, fa0);
        load_a_frag_lds(As, wm * 32 + 16, lane, fa1);
        load_b_frag_lds(Bs, wn * 32,      lane, fb0);
        load_b_frag_lds(Bs, wn * 32 + 16, lane, fb1);
        acc[0][0] = __builtin_amdgcn_wmma_f32_16x16x32_f16(false, fa0.h, false, fb0.h,
                                                           (short)0, acc[0][0], false, false);
        acc[0][1] = __builtin_amdgcn_wmma_f32_16x16x32_f16(false, fa0.h, false, fb1.h,
                                                           (short)0, acc[0][1], false, false);
        acc[1][0] = __builtin_amdgcn_wmma_f32_16x16x32_f16(false, fa1.h, false, fb0.h,
                                                           (short)0, acc[1][0], false, false);
        acc[1][1] = __builtin_amdgcn_wmma_f32_16x16x32_f16(false, fa1.h, false, fb1.h,
                                                           (short)0, acc[1][1], false, false);
        __syncthreads();
    }

    // C layout (ISA 7.12.2): VGPR r -> M = r (+8 for lanes 16-31), N = lane&15
#pragma unroll
    for (int i = 0; i < 2; ++i) {
#pragma unroll
        for (int j = 0; j < 2; ++j) {
#pragma unroll
            for (int r = 0; r < 8; ++r) {
                int m = m0 + wm * 32 + i * 16 + ((lane < 16) ? 0 : 8) + r;
                int n = n0 + wn * 32 + j * 16 + (lane & 15);
                Cb[(long)m * N + n] = acc[i][j][r];
            }
        }
    }
}

// ===========================================================================
// K2: depthwise 3x3, pad=1.  [B*576][128][128] fp32, 4 pixels per thread.
// ===========================================================================
__global__ __launch_bounds__(256)
void dwconv3x3(const float* __restrict__ in, const float* __restrict__ w,
               float* __restrict__ out) {
    long gid = (long)blockIdx.x * 256 + threadIdx.x;   // quad index
    int q = (int)(gid & 4095);                          // 128 rows * 32 quads
    long bc = gid >> 12;
    int ch = (int)(bc % 576);
    int y = q >> 5, x0 = (q & 31) << 2;
    const float* wp = w + ch * 9;
    const float* ip = in + bc * 16384;

    float a0 = 0.f, a1 = 0.f, a2 = 0.f, a3 = 0.f;
#pragma unroll
    for (int dy = -1; dy <= 1; ++dy) {
        int yy = y + dy;
        if (yy < 0 || yy >= 128) continue;
        const float* row = ip + yy * 128 + x0;
        float4 c = *(const float4*)row;
        float lft = (x0 > 0)   ? row[-1] : 0.f;
        float rgt = (x0 < 124) ? row[4]  : 0.f;
        float w0 = wp[(dy + 1) * 3 + 0];
        float w1 = wp[(dy + 1) * 3 + 1];
        float w2 = wp[(dy + 1) * 3 + 2];
        a0 += lft * w0 + c.x * w1 + c.y * w2;
        a1 += c.x * w0 + c.y * w1 + c.z * w2;
        a2 += c.y * w0 + c.z * w1 + c.w * w2;
        a3 += c.z * w0 + c.w * w1 + rgt * w2;
    }
    *(float4*)(out + bc * 16384 + y * 128 + x0) = float4{a0, a1, a2, a3};
}

// ===========================================================================
// K3: inverse L2 norm over L=16384 for q (ch 0..191) and k (ch 192..383).
// ===========================================================================
__global__ __launch_bounds__(256)
void norms_kernel(const float* __restrict__ qkv, float* __restrict__ norms) {
    constexpr int L = 16384;
    __shared__ float s[256];
    int idx = blockIdx.x;              // 0 .. 8*384-1
    int b = idx / 384, c = idx % 384;  // qkv channels 0..383 are exactly q|k
    const float4* p4 = (const float4*)(qkv + ((long)b * 576 + c) * L);
    float ss = 0.f;
    for (int i = threadIdx.x; i < L / 4; i += 256) {
        float4 v = p4[i];
        ss += v.x * v.x + v.y * v.y + v.z * v.z + v.w * v.w;
    }
    s[threadIdx.x] = ss;
    __syncthreads();
    for (int off = 128; off > 0; off >>= 1) {
        if (threadIdx.x < off) s[threadIdx.x] += s[threadIdx.x + off];
        __syncthreads();
    }
    if (threadIdx.x == 0) norms[idx] = 1.0f / fmaxf(sqrtf(s[0]), 1e-12f);
}

__global__ __launch_bounds__(256)
void zero_kernel(float* __restrict__ p, int n) {
    int i = blockIdx.x * 256 + threadIdx.x;
    if (i < n) p[i] = 0.f;
}

// ===========================================================================
// K4: attn_raw[b,h] += qhat(48,Kchunk) @ khat^T.  grid=(KSPLIT=64, B*H=32).
// Fragments gathered from global with b128 loads (f16 K-pairs {0..7},{16..23}
// are contiguous), scaled by inverse norms, packed-converted to f16.
// LDS ds_add_f32 reduce across waves, then global atomic add.
// ===========================================================================
__global__ __launch_bounds__(256)
void attn_partial(const float* __restrict__ qkv, const float* __restrict__ norms,
                  float* __restrict__ attn_raw) {
    constexpr int L = 16384;
    __shared__ float red[48 * 48];
    const int tid = threadIdx.x, lane = tid & 31, wave = tid >> 5;
    const int bh = blockIdx.y;
    const int b = bh >> 2, h = bh & 3;
    const int l0 = blockIdx.x * 256 + wave * 32;
    const float* base = qkv + (long)b * 576 * L;
    const float* nrm  = norms + b * 384;

    for (int i = tid; i < 48 * 48; i += 256) red[i] = 0.f;
    __syncthreads();

    const int mr  = lane & 15;
    const int kbA = (lane < 16) ? 0 : 8;
    const int kbB = (lane < 16) ? 0 : 8;

    AFrag a[3], bf[3];
#pragma unroll
    for (int t = 0; t < 3; ++t) {
        int ch = h * 48 + t * 16 + mr;            // 0..191
        float sq = nrm[ch];                       // q inv-norm
        const float4* pq4 = (const float4*)(base + (long)ch * L + l0 + kbA);
        float4 v0 = pq4[0], v1 = pq4[1];          // K = kb+0..7
        float4 v2 = pq4[4], v3 = pq4[5];          // K = kb+16..23
        a[t].u[0] = pk2(v0.x, v0.y, sq); a[t].u[1] = pk2(v0.z, v0.w, sq);
        a[t].u[2] = pk2(v1.x, v1.y, sq); a[t].u[3] = pk2(v1.z, v1.w, sq);
        a[t].u[4] = pk2(v2.x, v2.y, sq); a[t].u[5] = pk2(v2.z, v2.w, sq);
        a[t].u[6] = pk2(v3.x, v3.y, sq); a[t].u[7] = pk2(v3.z, v3.w, sq);

        float sk = nrm[192 + ch];                 // k inv-norm
        const float4* pk4 = (const float4*)(base + (long)(192 + ch) * L + l0 + kbB);
        float4 w0 = pk4[0], w1 = pk4[1], w2 = pk4[4], w3 = pk4[5];  // K = kb+0..7, kb+16..23
        bf[t].u[0] = pk2(w0.x, w0.y, sk); bf[t].u[1] = pk2(w0.z, w0.w, sk);
        bf[t].u[2] = pk2(w1.x, w1.y, sk); bf[t].u[3] = pk2(w1.z, w1.w, sk);
        bf[t].u[4] = pk2(w2.x, w2.y, sk); bf[t].u[5] = pk2(w2.z, w2.w, sk);
        bf[t].u[6] = pk2(w3.x, w3.y, sk); bf[t].u[7] = pk2(w3.z, w3.w, sk);
    }

#pragma unroll
    for (int mt = 0; mt < 3; ++mt) {
#pragma unroll
        for (int nt = 0; nt < 3; ++nt) {
            v8f acc = {};
            acc = __builtin_amdgcn_wmma_f32_16x16x32_f16(false, a[mt].h, false, bf[nt].h,
                                                         (short)0, acc, false, false);
#pragma unroll
            for (int r = 0; r < 8; ++r) {
                int m = mt * 16 + ((lane < 16) ? 0 : 8) + r;
                int n = nt * 16 + (lane & 15);
                atomicAdd(&red[m * 48 + n], acc[r]);   // ds_add_f32
            }
        }
    }
    __syncthreads();
    float* dst = attn_raw + (long)bh * 48 * 48;
    for (int i = tid; i < 48 * 48; i += 256) atomicAdd(&dst[i], red[i]);
}

// ===========================================================================
// K5: softmax(attn/sqrt(48) + pos_emb[h]) then M[b] = W_out @ blockdiag(attn).
// ===========================================================================
__global__ __launch_bounds__(256)
void softmax_mbuild(const float* __restrict__ attn_raw, const float* __restrict__ pos_emb,
                    const float* __restrict__ w_out, float* __restrict__ Mmat) {
    __shared__ float As[4 * 48 * 48];
    const int b = blockIdx.x, tid = threadIdx.x;
    const float scale = 0.14433756729740643f;   // 1/sqrt(48)
    if (tid < 192) {
        int h = tid / 48, c = tid % 48;
        const float* row = attn_raw + (((long)b * 4 + h) * 48 + c) * 48;
        float pe = pos_emb[h];
        float tmp[48];
        float mx = -1e30f;
#pragma unroll
        for (int d = 0; d < 48; ++d) { float v = row[d] * scale + pe; tmp[d] = v; mx = fmaxf(mx, v); }
        float sum = 0.f;
#pragma unroll
        for (int d = 0; d < 48; ++d) { float e = __expf(tmp[d] - mx); tmp[d] = e; sum += e; }
        float inv = 1.f / sum;
#pragma unroll
        for (int d = 0; d < 48; ++d) As[(h * 48 + c) * 48 + d] = tmp[d] * inv;
    }
    __syncthreads();
    for (int idx = tid; idx < 192 * 192; idx += 256) {
        int co = idx / 192, cin = idx % 192;
        int h = cin / 48, d = cin % 48;
        const float* wrow = w_out + co * 192 + h * 48;
        float s = 0.f;
#pragma unroll
        for (int cp = 0; cp < 48; ++cp) s += wrow[cp] * As[(h * 48 + cp) * 48 + d];
        Mmat[((long)b * 192 + co) * 192 + cin] = s;
    }
}

// ===========================================================================
extern "C" void kernel_launch(void* const* d_in, const int* in_sizes, int n_in,
                              void* d_out, int out_size, void* d_ws, size_t ws_size,
                              hipStream_t stream) {
    const float* x       = (const float*)d_in[0];
    const float* w_proj1 = (const float*)d_in[1];
    const float* w_dw    = (const float*)d_in[2];
    const float* pos_emb = (const float*)d_in[3];
    const float* w_out   = (const float*)d_in[4];
    float* out = (float*)d_out;
    (void)in_sizes; (void)n_in; (void)out_size; (void)ws_size;

    constexpr long Bn = 8, C = 192, C3 = 576, L = 16384;

    float* qkv_pre = (float*)d_ws;                 // [8][576][16384]
    float* qkv     = qkv_pre + Bn * C3 * L;        // [8][576][16384]
    float* norms   = qkv + Bn * C3 * L;            // [8][384]
    float* attn    = norms + Bn * 384;             // [8][4][48][48]
    float* Mmat    = attn + Bn * 4 * 48 * 48;      // [8][192][192]

    // K1: qkv_pre = W1 @ x   (per batch)
    gemm_f16wmma<<<dim3((int)(L / 128), (int)(C3 / 64), (int)Bn), 256, 0, stream>>>(
        w_proj1, 0L, x, C * L, qkv_pre, C3 * L, (int)C3, (int)L, (int)C);

    // K2: depthwise 3x3 (4 px / thread)
    dwconv3x3<<<dim3((int)((Bn * C3 * L) / (256 * 4))), 256, 0, stream>>>(qkv_pre, w_dw, qkv);

    // K3: inverse L2 norms of q,k rows
    norms_kernel<<<dim3((int)(Bn * 384)), 256, 0, stream>>>(qkv, norms);

    // K4 prep: zero the attention accumulator (atomics target)
    zero_kernel<<<dim3((int)((Bn * 4 * 48 * 48 + 255) / 256)), 256, 0, stream>>>(
        attn, (int)(Bn * 4 * 48 * 48));

    // K4: attention logits (K split across 64 blocks)
    attn_partial<<<dim3(64, (int)(Bn * 4)), 256, 0, stream>>>(qkv, norms, attn);

    // K5: softmax + M = W_out @ blockdiag(attn)
    softmax_mbuild<<<dim3((int)Bn), 256, 0, stream>>>(attn, pos_emb, w_out, Mmat);

    // K6: out = M[b] @ v[b]   (v = qkv channels 384..575)
    gemm_f16wmma<<<dim3((int)(L / 128), (int)(C / 64), (int)Bn), 256, 0, stream>>>(
        Mmat, 192L * 192L, qkv + 384 * L, C3 * L, out, C * L, (int)C, (int)L, (int)C);
}
